// IndexPropagationQuantize_11166914969845
// MI455X (gfx1250) — matrix operand order repaired and checked by reading, not compiled, not run on hardware
//
#include <hip/hip_runtime.h>
#include <hip/hip_bf16.h>
#include <math.h>

// ---------------------------------------------------------------------------
// VQ quantizer for MI455X (gfx1250, wave32, WMMA).
//   d = ||z||^2 + ||e||^2 - 2 z.e   via bf16 hi/lo split WMMA (3 products)
//   pass1: online softmax (m, S, T) + argmax  -> per-row stats
//   pass2: normalized probs -> per-WG batch partials (deterministic reduce)
// The three split products accumulate into INDEPENDENT v8f chains so the
// WMMA pipeline is not serialized on a single accumulator RAW chain.
// ---------------------------------------------------------------------------

#define N_ROWS   16384      // 16 * 32 * 32
#define NE       8192
#define EDIM     256
#define BETA_W   0.25f
#define INV_TEMP 100.0f     // 1/0.01
#define EPS_F    1e-10f
#define NEG_BIG  (-3.0e38f)

typedef __attribute__((ext_vector_type(16))) __bf16 v16bf;
typedef __attribute__((ext_vector_type(8)))  __bf16 v8bf;
typedef __attribute__((ext_vector_type(8)))  float  v8f;

union BF16x16 { v16bf v; v8bf h2[2]; };

struct AReg { BF16x16 hi[8]; BF16x16 lo[8]; };

// A (16x32 bf16) per-lane layout: lane = {M = lane&15, h = lane>>4};
// element j: k = s*32 + (j/8)*16 + h*8 + (j%8)  -> two contiguous 8-elem runs.
__device__ __forceinline__ void load_A(AReg& A, const __bf16* lzh,
                                       const __bf16* lzl, int lane) {
  const int M = lane & 15, h = lane >> 4;
#pragma unroll
  for (int s = 0; s < 8; ++s) {
    const int k0 = s * 32 + h * 8;
    A.hi[s].h2[0] = *(const v8bf*)(lzh + M * EDIM + k0);
    A.hi[s].h2[1] = *(const v8bf*)(lzh + M * EDIM + k0 + 16);
    A.lo[s].h2[0] = *(const v8bf*)(lzl + M * EDIM + k0);
    A.lo[s].h2[1] = *(const v8bf*)(lzl + M * EDIM + k0 + 16);
  }
}

// B (32x16 bf16): lane = {n = lane&15, g = lane>>4}; element j: k = (j/8)*16 + g*8 + (j%8)
// -> two contiguous 8-elem runs per lane per k-step (16B vector loads, L2 resident).
// Three independent accumulator chains (hi*hi, hi*lo, lo*hi).
__device__ __forceinline__ v8f gemm_tile(const AReg& A,
                                         const __bf16* __restrict__ ehi,
                                         const __bf16* __restrict__ elo,
                                         int tile, int lane) {
  v8f acc_hh = {};
  v8f acc_hl = {};
  v8f acc_lh = {};
  const int n = tile * 16 + (lane & 15);
  const int g = lane >> 4;
  const __bf16* bh = ehi + (size_t)n * EDIM + g * 8;
  const __bf16* bl = elo + (size_t)n * EDIM + g * 8;
#pragma unroll
  for (int s = 0; s < 8; ++s) {
    BF16x16 bhu, blu;
    bhu.h2[0] = *(const v8bf*)(bh + s * 32);
    bhu.h2[1] = *(const v8bf*)(bh + s * 32 + 16);
    blu.h2[0] = *(const v8bf*)(bl + s * 32);
    blu.h2[1] = *(const v8bf*)(bl + s * 32 + 16);
    v16bf bhv = bhu.v;
    v16bf blv = blu.v;
    acc_hh = __builtin_amdgcn_wmma_f32_16x16x32_bf16(
        false, A.hi[s].v, false, bhv, (short)0, acc_hh, false, false);
    acc_hl = __builtin_amdgcn_wmma_f32_16x16x32_bf16(
        false, A.hi[s].v, false, blv, (short)0, acc_hl, false, false);
    acc_lh = __builtin_amdgcn_wmma_f32_16x16x32_bf16(
        false, A.lo[s].v, false, bhv, (short)0, acc_lh, false, false);
  }
  return acc_hh + acc_hl + acc_lh;
}

__device__ __forceinline__ void softmax_merge(float& m, float& S, float& T,
                                              int& idx, float m2, float S2,
                                              float T2, int i2) {
  const float mn = fmaxf(m, m2);
  const float e1 = __expf(m - mn);
  const float e2 = __expf(m2 - mn);
  T = e1 * (T + (m - mn) * S) + e2 * (T2 + (m2 - mn) * S2);
  S = e1 * S + e2 * S2;
  if (m2 > m || (m2 == m && i2 < idx)) idx = i2;
  m = mn;
}

// ---------------------------------------------------------------------------
// Pack kernels: f32 -> bf16 hi/lo split + squared row norms.
// ---------------------------------------------------------------------------
__global__ void __launch_bounds__(256)
pack_z_kernel(const float* __restrict__ z, __bf16* __restrict__ zhi,
              __bf16* __restrict__ zlo, float* __restrict__ znorm) {
  __shared__ float sred[256];
  const int row = blockIdx.x, c = threadIdx.x;
  const int b = row >> 10, hw = row & 1023;
  const float x = z[((size_t)(b * EDIM + c) << 10) + hw];
  const __bf16 hb = (__bf16)x;
  const float hf = (float)hb;
  zhi[(size_t)row * EDIM + c] = hb;
  zlo[(size_t)row * EDIM + c] = (__bf16)(x - hf);
  sred[c] = x * x;
  __syncthreads();
  for (int s = 128; s > 0; s >>= 1) {
    if (c < s) sred[c] += sred[c + s];
    __syncthreads();
  }
  if (c == 0) znorm[row] = sred[0];
}

__global__ void __launch_bounds__(256)
pack_e_kernel(const float* __restrict__ e, __bf16* __restrict__ ehi,
              __bf16* __restrict__ elo, float* __restrict__ enorm) {
  __shared__ float sred[256];
  const int row = blockIdx.x, c = threadIdx.x;
  const float x = e[(size_t)row * EDIM + c];
  const __bf16 hb = (__bf16)x;
  const float hf = (float)hb;
  ehi[(size_t)row * EDIM + c] = hb;
  elo[(size_t)row * EDIM + c] = (__bf16)(x - hf);
  sred[c] = x * x;
  __syncthreads();
  for (int s = 128; s > 0; s >>= 1) {
    if (c < s) sred[c] += sred[c + s];
    __syncthreads();
  }
  if (c == 0) enorm[row] = sred[0];
}

// ---------------------------------------------------------------------------
// Pass 1: WMMA GEMM sweep + online softmax (m, S, T) + argmax per row.
// Block = 16 rows, 8 waves; wave w handles column tiles w, w+8, ...
// ---------------------------------------------------------------------------
__global__ void __launch_bounds__(256)
vq_pass1_kernel(const __bf16* __restrict__ zhi, const __bf16* __restrict__ zlo,
                const float* __restrict__ znorm,
                const __bf16* __restrict__ ehi, const __bf16* __restrict__ elo,
                const float* __restrict__ enorm,
                float* __restrict__ row_m, float* __restrict__ row_S,
                float* __restrict__ row_H, int* __restrict__ row_idx) {
  __shared__ __align__(32) __bf16 lzh[16 * EDIM];
  __shared__ __align__(32) __bf16 lzl[16 * EDIM];
  __shared__ float lzn[16];
  __shared__ float red_m[8][16], red_S[8][16], red_T[8][16];
  __shared__ int red_i[8][16];

  const int t = threadIdx.x;
  const int wave = t >> 5, lane = t & 31, g = lane >> 4;
  const int row0 = blockIdx.x * 16;

  ((v16bf*)lzh)[t] = ((const v16bf*)(zhi + (size_t)row0 * EDIM))[t];
  ((v16bf*)lzl)[t] = ((const v16bf*)(zlo + (size_t)row0 * EDIM))[t];
  if (t < 16) lzn[t] = znorm[row0 + t];
  __syncthreads();

  AReg A;
  load_A(A, lzh, lzl, lane);

  float zn[8], m[8], S[8], T[8];
  int idx[8];
#pragma unroll
  for (int r = 0; r < 8; ++r) {
    zn[r] = lzn[g * 8 + r];
    m[r] = NEG_BIG;
    S[r] = 0.0f;
    T[r] = 0.0f;
    idx[r] = 0;
  }

  for (int tile = wave; tile < NE / 16; tile += 8) {
    v8f acc = gemm_tile(A, ehi, elo, tile, lane);
    const int n = tile * 16 + (lane & 15);
    const float en = enorm[n];
#pragma unroll
    for (int r = 0; r < 8; ++r) {
      const float dist = zn[r] + en - 2.0f * acc[r];
      const float l = -dist * INV_TEMP;
      if (l > m[r]) {
        const float e1 = __expf(m[r] - l);
        T[r] = e1 * (T[r] + (m[r] - l) * S[r]);
        S[r] = e1 * S[r] + 1.0f;
        m[r] = l;
        idx[r] = n;
      } else {
        const float u = __expf(l - m[r]);
        S[r] += u;
        T[r] += u * (l - m[r]);
      }
    }
    if (tile + 8 < NE / 16)
      __builtin_prefetch(ehi + (size_t)(tile + 8) * 16 * EDIM, 0, 1);
  }

  // Reduce across the 16 column-lanes (same row, different n).
#pragma unroll
  for (int off = 1; off < 16; off <<= 1) {
#pragma unroll
    for (int r = 0; r < 8; ++r) {
      const float m2 = __shfl_xor(m[r], off, 32);
      const float S2 = __shfl_xor(S[r], off, 32);
      const float T2 = __shfl_xor(T[r], off, 32);
      const int i2 = __shfl_xor(idx[r], off, 32);
      softmax_merge(m[r], S[r], T[r], idx[r], m2, S2, T2, i2);
    }
  }
  if ((lane & 15) == 0) {
#pragma unroll
    for (int r = 0; r < 8; ++r) {
      const int row = g * 8 + r;
      red_m[wave][row] = m[r];
      red_S[wave][row] = S[r];
      red_T[wave][row] = T[r];
      red_i[wave][row] = idx[r];
    }
  }
  __syncthreads();
  if (t < 16) {
    float fm = red_m[0][t], fS = red_S[0][t], fT = red_T[0][t];
    int fi = red_i[0][t];
    for (int w = 1; w < 8; ++w)
      softmax_merge(fm, fS, fT, fi, red_m[w][t], red_S[w][t], red_T[w][t],
                    red_i[w][t]);
    row_m[row0 + t] = fm;
    row_S[row0 + t] = fS;
    row_H[row0 + t] = logf(fS) - fT / fS;  // H = log S - E[l - m]
    row_idx[row0 + t] = fi;
  }
}

// ---------------------------------------------------------------------------
// Pass 2: second GEMM sweep, normalized probs -> per-WG batch partials.
// ---------------------------------------------------------------------------
__global__ void __launch_bounds__(256)
vq_pass2_kernel(const __bf16* __restrict__ zhi, const __bf16* __restrict__ zlo,
                const float* __restrict__ znorm,
                const __bf16* __restrict__ ehi, const __bf16* __restrict__ elo,
                const float* __restrict__ enorm,
                const float* __restrict__ row_m, const float* __restrict__ row_S,
                float* __restrict__ batch_part) {
  __shared__ __align__(32) __bf16 lzh[16 * EDIM];
  __shared__ __align__(32) __bf16 lzl[16 * EDIM];
  __shared__ float lzn[16];

  const int t = threadIdx.x;
  const int wave = t >> 5, lane = t & 31, g = lane >> 4;
  const int row0 = blockIdx.x * 16;

  ((v16bf*)lzh)[t] = ((const v16bf*)(zhi + (size_t)row0 * EDIM))[t];
  ((v16bf*)lzl)[t] = ((const v16bf*)(zlo + (size_t)row0 * EDIM))[t];
  if (t < 16) lzn[t] = znorm[row0 + t];
  __syncthreads();

  AReg A;
  load_A(A, lzh, lzl, lane);

  float zn[8], mr[8], si[8];
#pragma unroll
  for (int r = 0; r < 8; ++r) {
    const int row = row0 + g * 8 + r;
    zn[r] = lzn[g * 8 + r];
    mr[r] = row_m[row];
    si[r] = 1.0f / row_S[row];
  }

  float* out = batch_part + (size_t)blockIdx.x * NE;
  for (int tile = wave; tile < NE / 16; tile += 8) {
    v8f acc = gemm_tile(A, ehi, elo, tile, lane);
    const int n = tile * 16 + (lane & 15);
    const float en = enorm[n];
    float ps = 0.0f;
#pragma unroll
    for (int r = 0; r < 8; ++r) {
      const float l = -(zn[r] + en - 2.0f * acc[r]) * INV_TEMP;
      ps += __expf(l - mr[r]) * si[r];
    }
    ps += __shfl_xor(ps, 16, 32);  // combine rows 0-7 with 8-15 (same n)
    if (g == 0) out[n] = ps;
    if (tile + 8 < NE / 16)
      __builtin_prefetch(ehi + (size_t)(tile + 8) * 16 * EDIM, 0, 1);
  }
}

// Deterministic sequential reduce of per-WG batch partials -> p*log(p+eps) terms.
__global__ void __launch_bounds__(256)
batch_reduce_kernel(const float* __restrict__ batch_part,
                    float* __restrict__ bent) {
  const int c = blockIdx.x * blockDim.x + threadIdx.x;
  float s = 0.0f;
  for (int w = 0; w < N_ROWS / 16; ++w) s += batch_part[(size_t)w * NE + c];
  const float p = s * (1.0f / (float)N_ROWS);
  bent[c] = p * logf(p + EPS_F);
}

// Gather z_q = embedding[idx], emit transposed output, per-row MSE partials,
// and the index output (as float, per harness output dtype).
__global__ void __launch_bounds__(256)
gather_kernel(const float* __restrict__ z, const float* __restrict__ emb,
              const int* __restrict__ row_idx, float* __restrict__ out,
              float* __restrict__ row_mse) {
  __shared__ float sred[256];
  const int row = blockIdx.x, c = threadIdx.x;
  const int b = row >> 10, hw = row & 1023;
  const int id = row_idx[row];
  const float ev = emb[(size_t)id * EDIM + c];
  const size_t zi = ((size_t)(b * EDIM + c) << 10) + hw;
  const float zv = z[zi];
  out[zi] = ev;  // straight-through output == z_q numerically
  const float d = ev - zv;
  sred[c] = d * d;
  __syncthreads();
  for (int s = 128; s > 0; s >>= 1) {
    if (c < s) sred[c] += sred[c + s];
    __syncthreads();
  }
  if (c == 0) {
    row_mse[row] = sred[0];
    out[(size_t)N_ROWS * EDIM + 5 + row] = (float)id;
  }
}

__global__ void __launch_bounds__(256)
hist_kernel(const int* __restrict__ row_idx, int* __restrict__ hist) {
  const int i = blockIdx.x * blockDim.x + threadIdx.x;
  if (i < N_ROWS) atomicAdd(&hist[row_idx[i]], 1);
}

__global__ void __launch_bounds__(256)
finalize_kernel(const float* __restrict__ row_H, const float* __restrict__ row_mse,
                const float* __restrict__ bent, const int* __restrict__ hist,
                float* __restrict__ out) {
  __shared__ float sred[256];
  const int t = threadIdx.x;
  float acc[4] = {0.f, 0.f, 0.f, 0.f};
  for (int i = t; i < N_ROWS; i += 256) acc[0] += row_H[i];
  for (int i = t; i < N_ROWS; i += 256) acc[1] += row_mse[i];
  for (int i = t; i < NE; i += 256) acc[2] += bent[i];
  for (int i = t; i < NE; i += 256) {
    const float em = (float)hist[i] * (1.0f / (float)N_ROWS);
    acc[3] += em * logf(em + EPS_F);
  }
  float res[4];
  for (int k = 0; k < 4; ++k) {
    sred[t] = acc[k];
    __syncthreads();
    for (int s = 128; s > 0; s >>= 1) {
      if (t < s) sred[t] += sred[t + s];
      __syncthreads();
    }
    res[k] = sred[0];
    __syncthreads();
  }
  if (t == 0) {
    const float avgH = res[0] / (float)N_ROWS;
    const float mse = res[1] / (float)((size_t)N_ROWS * EDIM);
    const float loss = (1.0f + BETA_W) * mse;  // both terms equal numerically
    const float bH = -res[2];
    const float eloss = avgH - bH;  // SAMPLE_MIN_W=BATCH_MAX_W=1
    const float perp = __expf(-res[3]);
    float* s = out + (size_t)N_ROWS * EDIM;
    s[0] = loss;
    s[1] = avgH;
    s[2] = bH;
    s[3] = eloss;
    s[4] = perp;
  }
}

// ---------------------------------------------------------------------------
extern "C" void kernel_launch(void* const* d_in, const int* in_sizes, int n_in,
                              void* d_out, int out_size, void* d_ws,
                              size_t ws_size, hipStream_t stream) {
  const float* z = (const float*)d_in[0];
  const float* emb = (const float*)d_in[1];
  float* out = (float*)d_out;
  char* ws = (char*)d_ws;

  size_t off = 0;
  auto carve = [&](size_t bytes) -> char* {
    char* p = ws + off;
    off = (off + bytes + 255) & ~(size_t)255;
    return p;
  };

  __bf16* zhi = (__bf16*)carve((size_t)N_ROWS * EDIM * 2);
  __bf16* zlo = (__bf16*)carve((size_t)N_ROWS * EDIM * 2);
  float* znorm = (float*)carve((size_t)N_ROWS * 4);
  __bf16* ehi = (__bf16*)carve((size_t)NE * EDIM * 2);
  __bf16* elo = (__bf16*)carve((size_t)NE * EDIM * 2);
  float* enorm = (float*)carve((size_t)NE * 4);
  float* row_m = (float*)carve((size_t)N_ROWS * 4);
  float* row_S = (float*)carve((size_t)N_ROWS * 4);
  float* row_H = (float*)carve((size_t)N_ROWS * 4);
  int* row_idx = (int*)carve((size_t)N_ROWS * 4);
  float* row_mse = (float*)carve((size_t)N_ROWS * 4);
  float* bent = (float*)carve((size_t)NE * 4);
  int* hist = (int*)carve((size_t)NE * 4);
  float* batch_part = (float*)carve((size_t)(N_ROWS / 16) * NE * 4);  // 32 MB

  pack_z_kernel<<<N_ROWS, 256, 0, stream>>>(z, zhi, zlo, znorm);
  pack_e_kernel<<<NE, 256, 0, stream>>>(emb, ehi, elo, enorm);
  hipMemsetAsync(hist, 0, (size_t)NE * 4, stream);

  vq_pass1_kernel<<<N_ROWS / 16, 256, 0, stream>>>(
      zhi, zlo, znorm, ehi, elo, enorm, row_m, row_S, row_H, row_idx);
  vq_pass2_kernel<<<N_ROWS / 16, 256, 0, stream>>>(
      zhi, zlo, znorm, ehi, elo, enorm, row_m, row_S, batch_part);
  batch_reduce_kernel<<<NE / 256, 256, 0, stream>>>(batch_part, bent);
  gather_kernel<<<N_ROWS, 256, 0, stream>>>(z, emb, row_idx, out, row_mse);
  hist_kernel<<<N_ROWS / 256, 256, 0, stream>>>(row_idx, hist);
  finalize_kernel<<<1, 256, 0, stream>>>(row_H, row_mse, bent, hist, out);
}
